// CondConv2d_5634997092874
// MI455X (gfx1250) — compile-verified, matrix operand order checked by reading, not run on hardware
//
#include <hip/hip_runtime.h>
#include <hip/hip_bf16.h>

typedef __attribute__((ext_vector_type(16))) _Float16 v16h;
typedef __attribute__((ext_vector_type(8)))  float    v8f;
typedef __attribute__((ext_vector_type(2)))  _Float16 half2_t;

#define BB 16
#define LL 8
#define CC 64
#define HH 50
#define WW 50
#define NG (BB*LL)          // 128 groups
#define NSP (HH*WW)         // 2500 spatial
#define KTOT (CC*9)         // 576
#define NT 64               // N tile per workgroup
#define KC 288              // K chunk = 32 channels * 9
#define XSTRIDE 296         // padded LDS row stride (f16 elems)

// ---------------- Kernel 1: adaptive max pool 50x50 -> 5x5 ----------------
__global__ __launch_bounds__(256) void pool_kernel(const float* __restrict__ x,
                                                   float* __restrict__ pooled) {
    int gid = blockIdx.x * 256 + threadIdx.x;          // 16*8*64*25 = 204800
    int s  = gid % 25;
    int c  = (gid / 25) % CC;
    int g  = gid / (25 * CC);                          // g = b*8 + l
    int py = s / 5, px = s % 5;
    const float* src = x + ((size_t)g * CC + c) * NSP;
    float m = -3.4e38f;
    for (int y = py * 10; y < py * 10 + 10; ++y)
        for (int xx = px * 10; xx < px * 10 + 10; ++xx) {
            float v = src[y * WW + xx];
            m = v > m ? v : m;
        }
    pooled[((size_t)g * CC + c) * 25 + s] = m;
}

// 3x3 conv (pad 1) on a 5x5 map held in LDS, Cin=64
__device__ __forceinline__ float conv5x5_at(const float* src, const float* __restrict__ w,
                                            float bias, int o, int n) {
    int y = n / 5, x = n % 5;
    float acc = bias;
    for (int i = 0; i < CC; ++i) {
        const float* wi = w + (o * CC + i) * 9;
        const float* si = src + i * 25;
        #pragma unroll
        for (int dy = -1; dy <= 1; ++dy) {
            int yy = y + dy;
            if (yy < 0 || yy > 4) continue;
            #pragma unroll
            for (int dx = -1; dx <= 1; ++dx) {
                int xx = x + dx;
                if (xx < 0 || xx > 4) continue;
                acc += wi[(dy + 1) * 3 + (dx + 1)] * si[yy * 5 + xx];
            }
        }
    }
    return acc;
}

// ---------------- Kernel 2: prior conv + sequential attention scan ----------------
__global__ __launch_bounds__(256) void scan_kernel(
    const float* __restrict__ pooled, const float* __restrict__ init_w,
    const float* __restrict__ init_b, const float* __restrict__ q_w,
    const float* __restrict__ q_b, const float* __restrict__ k_w,
    const float* __restrict__ k_b, const float* __restrict__ v_w,
    const float* __restrict__ v_b, float* __restrict__ avg) {
    __shared__ float prior[CC * 25];
    __shared__ float xt[CC * 25];
    __shared__ float qbuf[4 * 25];
    __shared__ float kbuf[4 * 25];
    __shared__ float att[25 * 25];
    __shared__ float vbuf[CC * 25];
    const int b = blockIdx.x;
    const int tid = threadIdx.x;

    for (int e = tid; e < CC * 25; e += 256)
        xt[e] = pooled[(size_t)(b * LL + 0) * CC * 25 + e];
    __syncthreads();
    for (int e = tid; e < CC * 25; e += 256)
        prior[e] = conv5x5_at(xt, init_w, init_b[e / 25], e / 25, e % 25);
    __syncthreads();

    for (int l = 0; l < LL; ++l) {
        for (int e = tid; e < CC * 25; e += 256)
            xt[e] = pooled[(size_t)(b * LL + l) * CC * 25 + e];
        __syncthreads();
        if (tid < 100) {                      // q from prior
            int o = tid / 25, n = tid % 25;
            qbuf[tid] = conv5x5_at(prior, q_w, q_b[o], o, n);
        } else if (tid < 200) {               // k from xt
            int t = tid - 100;
            int o = t / 25, n = t % 25;
            kbuf[t] = conv5x5_at(xt, k_w, k_b[o], o, n);
        }
        for (int e = tid; e < CC * 25; e += 256)   // v from xt
            vbuf[e] = conv5x5_at(xt, v_w, v_b[e / 25], e / 25, e % 25);
        __syncthreads();
        for (int e = tid; e < 625; e += 256) {
            int n = e / 25, m = e % 25;
            float s = 0.f;
            #pragma unroll
            for (int ch = 0; ch < 4; ++ch) s += qbuf[ch * 25 + n] * kbuf[ch * 25 + m];
            att[e] = s;
        }
        __syncthreads();
        if (tid < 25) {                       // softmax over m
            float mx = -3.4e38f;
            for (int m = 0; m < 25; ++m) mx = att[tid * 25 + m] > mx ? att[tid * 25 + m] : mx;
            float sum = 0.f;
            for (int m = 0; m < 25; ++m) { float e2 = __expf(att[tid * 25 + m] - mx); att[tid * 25 + m] = e2; sum += e2; }
            float inv = 1.f / sum;
            for (int m = 0; m < 25; ++m) att[tid * 25 + m] *= inv;
        }
        __syncthreads();
        for (int e = tid; e < CC * 25; e += 256) { // new prior = V * attn^T
            int c = e / 25, n = e % 25;
            float s = 0.f;
            for (int m = 0; m < 25; ++m) s += vbuf[c * 25 + m] * att[n * 25 + m];
            prior[e] = s;
        }
        __syncthreads();
        if (tid < CC) {                       // spatial mean
            float s = 0.f;
            for (int m = 0; m < 25; ++m) s += prior[tid * 25 + m];
            avg[((size_t)b * CC + tid) * LL + l] = s * (1.0f / 25.0f);
        }
        __syncthreads();
    }
}

// ---------------- Kernel 3: calibration (scale, fc) ----------------
__global__ __launch_bounds__(256) void calib_kernel(
    const float* __restrict__ avg, const float* __restrict__ tconv_w,
    const float* __restrict__ tconv_b, const float* __restrict__ fc_w,
    const float* __restrict__ fc_b, float* __restrict__ scale,
    float* __restrict__ fcb) {
    int gid = blockIdx.x * 256 + threadIdx.x;     // 8192
    int g = gid >> 6, c = gid & 63;
    int b = g >> 3, l = g & 7;
    float s = tconv_b[c];
    for (int i = 0; i < CC; ++i)
        s += tconv_w[c * CC + i] * avg[((size_t)b * CC + i) * LL + l];
    scale[gid] = 1.0f + s;
    if (gid < NG) {
        int bb = gid >> 3, ll = gid & 7;
        float f = fc_b[0];
        for (int i = 0; i < CC; ++i)
            f += fc_w[i] * avg[((size_t)bb * CC + i) * LL + ll];
        fcb[gid] = 1.0f + f;
    }
}

// ---------------- Kernel 4: scaled f16 weights + bias ----------------
__global__ __launch_bounds__(256) void wprep_kernel(
    const float* __restrict__ weight, const float* __restrict__ bias,
    const float* __restrict__ scale, const float* __restrict__ fcb,
    _Float16* __restrict__ fw16, float* __restrict__ fb) {
    long gid = (long)blockIdx.x * 256 + threadIdx.x;     // 128*64*576
    if (gid >= (long)NG * CC * KTOT) return;
    int kk = (int)(gid % KTOT);
    int o  = (int)((gid / KTOT) % CC);
    int g  = (int)(gid / ((long)KTOT * CC));
    int i  = kk / 9;
    fw16[gid] = (_Float16)(weight[o * KTOT + kk] * scale[g * CC + i]);
    if (kk == 0) fb[g * CC + o] = bias[o] * fcb[g];
}

// ---------------- Kernel 5: grouped conv as implicit GEMM via WMMA ----------------
// D[64 x 64-tile] = W16[64 x 576] * im2col(X)[576 x 64-tile], f32 accumulate.
__global__ __launch_bounds__(256) void conv_main(
    const float* __restrict__ x, const _Float16* __restrict__ fw16,
    const float* __restrict__ fb, float* __restrict__ out) {
    __shared__ _Float16 Xlds[NT * XSTRIDE];              // 37888 B
    const int g     = blockIdx.y;                        // 0..127
    const int ntile = blockIdx.x;                        // 0..39
    const int tid   = threadIdx.x;
    const int lane  = tid & 31;
    const int wave  = tid >> 5;                          // 0..7
    const int mi    = wave & 3;                          // M block (16 rows)
    const int ni    = wave >> 2;                         // 0..1 -> 32 cols
    const int n_base = ntile * NT;
    const float* xg = x + (size_t)g * CC * NSP;

    v8f acc0 = {}; v8f acc1 = {};
    const int arow  = lane & 15;
    const int khalf = (lane >> 4) << 3;                  // 0 or 8

    for (int cc = 0; cc < 2; ++cc) {
        __syncthreads();
        // stage im2col chunk: channels [cc*32, cc*32+32), K-rows kk=0..287
        for (int e = tid; e < NT * KC; e += 256) {
            int n  = e & (NT - 1);
            int kk = e >> 6;
            int c  = kk / 9;
            int r  = kk - c * 9;
            int dy = r / 3 - 1, dx = r % 3 - 1;
            int ng = n_base + n;
            float v = 0.f;
            if (ng < NSP) {
                int y  = ng / WW + dy;
                int xq = ng % WW + dx;
                if (y >= 0 && y < HH && xq >= 0 && xq < WW)
                    v = xg[(size_t)(cc * 32 + c) * NSP + y * WW + xq];
            }
            Xlds[n * XSTRIDE + kk] = (_Float16)v;
        }
        __syncthreads();

        const _Float16* wrow  = fw16 + ((size_t)g * CC + mi * 16 + arow) * KTOT + cc * KC;
        const _Float16* xrow0 = Xlds + (ni * 32 + arow) * XSTRIDE;
        const _Float16* xrow1 = xrow0 + 16 * XSTRIDE;
        __builtin_prefetch(wrow + KC, 0, 1);             // global_prefetch next chunk

        for (int kb = 0; kb < KC; kb += 32) {
            v16h a, b0, b1;
            #pragma unroll
            for (int v = 0; v < 8; ++v) {
                // 16-bit A/B fragment layout: lanes 0-15 K=0..7,16..23; lanes 16-31 K=8..15,24..31
                int k0 = ((v & 3) << 1) + khalf + ((v >> 2) << 4) + kb;
                half2_t ap  = *(const half2_t*)(wrow + k0);
                half2_t bp0 = *(const half2_t*)(xrow0 + k0);
                half2_t bp1 = *(const half2_t*)(xrow1 + k0);
                a [2 * v] = ap.x;  a [2 * v + 1] = ap.y;
                b0[2 * v] = bp0.x; b0[2 * v + 1] = bp0.y;
                b1[2 * v] = bp1.x; b1[2 * v + 1] = bp1.y;
            }
            acc0 = __builtin_amdgcn_wmma_f32_16x16x32_f16(false, a, false, b0,
                                                          (short)0, acc0, false, false);
            acc1 = __builtin_amdgcn_wmma_f32_16x16x32_f16(false, a, false, b1,
                                                          (short)0, acc1, false, false);
        }
    }

    // epilogue: C layout -> VGPR r: M = r + 8*(lane>=16), N = lane&15
    const int colL  = lane & 15;
    const int rhalf = (lane >> 4) << 3;
    float* og = out + (size_t)g * CC * NSP;
    #pragma unroll
    for (int r = 0; r < 8; ++r) {
        int o = mi * 16 + rhalf + r;
        float bb = fb[g * CC + o];
        int n0 = n_base + ni * 32 + colL;
        if (n0 < NSP) og[(size_t)o * NSP + n0] = acc0[r] + bb;
        int n1 = n0 + 16;
        if (n1 < NSP) og[(size_t)o * NSP + n1] = acc1[r] + bb;
    }
}

extern "C" void kernel_launch(void* const* d_in, const int* in_sizes, int n_in,
                              void* d_out, int out_size, void* d_ws, size_t ws_size,
                              hipStream_t stream) {
    const float* x       = (const float*)d_in[0];
    const float* weight  = (const float*)d_in[1];
    const float* bias    = (const float*)d_in[2];
    const float* init_w  = (const float*)d_in[3];
    const float* init_b  = (const float*)d_in[4];
    const float* q_w     = (const float*)d_in[5];
    const float* q_b     = (const float*)d_in[6];
    const float* k_w     = (const float*)d_in[7];
    const float* k_b     = (const float*)d_in[8];
    const float* v_w     = (const float*)d_in[9];
    const float* v_b     = (const float*)d_in[10];
    const float* tconv_w = (const float*)d_in[11];
    const float* tconv_b = (const float*)d_in[12];
    const float* fc_w    = (const float*)d_in[13];
    const float* fc_b    = (const float*)d_in[14];

    char* ws = (char*)d_ws;
    float*    pooled = (float*)(ws);                 // 819200 B
    float*    avg    = (float*)(ws + 819200);        // 32768 B
    float*    scale  = (float*)(ws + 851968);        // 32768 B
    float*    fcb    = (float*)(ws + 884736);        // 512 B
    float*    fb     = (float*)(ws + 885248);        // 32768 B
    _Float16* fw16   = (_Float16*)(ws + 918016);     // 9437184 B
    float*    out    = (float*)d_out;

    hipLaunchKernelGGL(pool_kernel, dim3(800), dim3(256), 0, stream, x, pooled);
    hipLaunchKernelGGL(scan_kernel, dim3(BB), dim3(256), 0, stream,
                       pooled, init_w, init_b, q_w, q_b, k_w, k_b, v_w, v_b, avg);
    hipLaunchKernelGGL(calib_kernel, dim3(32), dim3(256), 0, stream,
                       avg, tconv_w, tconv_b, fc_w, fc_b, scale, fcb);
    hipLaunchKernelGGL(wprep_kernel, dim3(18432), dim3(256), 0, stream,
                       weight, bias, scale, fcb, fw16, fb);
    hipLaunchKernelGGL(conv_main, dim3(40, NG), dim3(256), 0, stream,
                       x, fw16, fb, out);
}
